// AttentionLayer_8375186227604
// MI455X (gfx1250) — matrix operand order repaired and checked by reading, not compiled, not run on hardware
//
#include <hip/hip_runtime.h>
#include <hip/hip_bf16.h>

typedef __bf16 bf16t;
typedef __attribute__((ext_vector_type(16))) __bf16 v16bf;
typedef __attribute__((ext_vector_type(8)))  __bf16 v8bf;
typedef __attribute__((ext_vector_type(8)))  float  v8f;

#define BN_EPS 1e-5f

// ---------------------------------------------------------------------------
// 16-lane (DPP row) butterfly reductions, pure VALU — no LDS/bpermute.
// Combining network xor1 / xor2 / xor7 / xor15 touches every pair in a row,
// which is sufficient for commutative max / add.
// ---------------------------------------------------------------------------
#if __has_builtin(__builtin_amdgcn_update_dpp)
__device__ __forceinline__ float dpp_bfly(float x, int ctrl_dummy, int step)
{
    int v = __builtin_bit_cast(int, x);
    int t;
    switch (step) {
    case 0: t = __builtin_amdgcn_update_dpp(v, v, 0xB1,  0xF, 0xF, true); break; // quad_perm [1,0,3,2]
    case 1: t = __builtin_amdgcn_update_dpp(v, v, 0x4E,  0xF, 0xF, true); break; // quad_perm [2,3,0,1]
    case 2: t = __builtin_amdgcn_update_dpp(v, v, 0x141, 0xF, 0xF, true); break; // row_half_mirror
    default:t = __builtin_amdgcn_update_dpp(v, v, 0x140, 0xF, 0xF, true); break; // row_mirror
    }
    (void)ctrl_dummy;
    return __builtin_bit_cast(float, t);
}
__device__ __forceinline__ float row16_max(float x)
{
    x = fmaxf(x, dpp_bfly(x, 0, 0));
    x = fmaxf(x, dpp_bfly(x, 0, 1));
    x = fmaxf(x, dpp_bfly(x, 0, 2));
    x = fmaxf(x, dpp_bfly(x, 0, 3));
    return x;
}
__device__ __forceinline__ float row16_sum(float x)
{
    x += dpp_bfly(x, 0, 0);
    x += dpp_bfly(x, 0, 1);
    x += dpp_bfly(x, 0, 2);
    x += dpp_bfly(x, 0, 3);
    return x;
}
#else
__device__ __forceinline__ float row16_max(float x)
{
    x = fmaxf(x, __shfl_xor(x, 1, 16));
    x = fmaxf(x, __shfl_xor(x, 2, 16));
    x = fmaxf(x, __shfl_xor(x, 4, 16));
    x = fmaxf(x, __shfl_xor(x, 8, 16));
    return x;
}
__device__ __forceinline__ float row16_sum(float x)
{
    x += __shfl_xor(x, 1, 16);
    x += __shfl_xor(x, 2, 16);
    x += __shfl_xor(x, 4, 16);
    x += __shfl_xor(x, 8, 16);
    return x;
}
#endif

// ---------------------------------------------------------------------------
// 3x3 conv (pad=1, no conv bias) + BN(eval) + ReLU, NCHW fp32.
// ---------------------------------------------------------------------------
__global__ __launch_bounds__(256) void conv3x3_bn_relu_kernel(
    const float* __restrict__ x, const float* __restrict__ w,
    const float* __restrict__ bng, const float* __restrict__ bnb,
    const float* __restrict__ bnm, const float* __restrict__ bnv,
    float* __restrict__ y, int B, int Cin, int Cout, int H, int W)
{
    int idx = blockIdx.x * blockDim.x + threadIdx.x;
    int total = B * Cout * H * W;
    if (idx >= total) return;
    int wp = idx % W; int t = idx / W;
    int hp = t % H;   t /= H;
    int co = t % Cout; int b = t / Cout;

    const float* wbase = w + (size_t)co * Cin * 9;
    const float* xb    = x + (size_t)b * Cin * H * W;

    float acc = 0.f;
    for (int ci = 0; ci < Cin; ++ci) {
        const float* xc = xb + (size_t)ci * H * W;
        const float* wc = wbase + ci * 9;
        #pragma unroll
        for (int kh = 0; kh < 3; ++kh) {
            int hh = hp + kh - 1;
            if (hh < 0 || hh >= H) continue;
            const float* xr = xc + (size_t)hh * W;
            #pragma unroll
            for (int kw = 0; kw < 3; ++kw) {
                int ww = wp + kw - 1;
                if (ww < 0 || ww >= W) continue;
                acc = fmaf(xr[ww], wc[kh * 3 + kw], acc);
            }
        }
    }
    float sc = bng[co] * rsqrtf(bnv[co] + BN_EPS);
    float sh = bnb[co] - bnm[co] * sc;
    float r  = fmaf(acc, sc, sh);
    y[idx] = r > 0.f ? r : 0.f;
}

// ---------------------------------------------------------------------------
// Fused 1x1 q/k/v projections (with bias). Emits bf16 tensors laid out for
// direct WMMA operand loads:
//   qT,kT : [B][N][16]  (channel fastest)
//   v     : [B][64][N]  (pixel fastest)
// ---------------------------------------------------------------------------
__global__ __launch_bounds__(256) void qkv_kernel(
    const float* __restrict__ x1,
    const float* __restrict__ wq, const float* __restrict__ bq,
    const float* __restrict__ wk, const float* __restrict__ bk,
    const float* __restrict__ wv, const float* __restrict__ bv,
    bf16t* __restrict__ qT, bf16t* __restrict__ kT, bf16t* __restrict__ vbf,
    int B, int C, int Cqk, int N)
{
    int OCtot = 2 * Cqk + C;            // 96
    int idx = blockIdx.x * blockDim.x + threadIdx.x;
    int total = B * OCtot * N;
    if (idx >= total) return;
    int i  = idx % N; int t = idx / N;
    int oc = t % OCtot; int b = t / OCtot;

    const float* xb = x1 + (size_t)b * C * N + i;
    const float* wrow;
    float acc;
    if (oc < Cqk)            { wrow = wq + (size_t)oc * C;             acc = bq[oc]; }
    else if (oc < 2 * Cqk)   { wrow = wk + (size_t)(oc - Cqk) * C;     acc = bk[oc - Cqk]; }
    else                     { wrow = wv + (size_t)(oc - 2 * Cqk) * C; acc = bv[oc - 2 * Cqk]; }

    for (int c = 0; c < C; ++c)
        acc = fmaf(xb[(size_t)c * N], wrow[c], acc);

    bf16t h = (bf16t)acc;
    if (oc < Cqk)          qT[((size_t)b * N + i) * Cqk + oc] = h;
    else if (oc < 2 * Cqk) kT[((size_t)b * N + i) * Cqk + (oc - Cqk)] = h;
    else                   vbf[((size_t)b * C + (oc - 2 * Cqk)) * N + i] = h;
}

// ---------------------------------------------------------------------------
// Flash attention over N=9216 pixels, 16 q/k channels, 64 v channels.
// One wave32 per 16-row i-tile. j streamed in blocks of 32.
//   S(16x16) = A(qT tile 16x32, K padded) x B(kT tile 32x16)  via wmma bf16
//   O(16x16c) += A(P 16x32) x B(vT 32x16)                      via wmma bf16
// P transposed C-layout -> A-layout through wave-private LDS.
// Epilogue: feat = gamma * (O / l) + x1   (residual), fp32.
// ---------------------------------------------------------------------------
#define ATTN_WAVES 4

__global__ __launch_bounds__(128) void flash_attn_kernel(
    const bf16t* __restrict__ qT, const bf16t* __restrict__ kT,
    const bf16t* __restrict__ vbf, const float* __restrict__ x1,
    const float* __restrict__ gamma, float* __restrict__ feat,
    int Bn, int N, int C)
{
    __shared__ __align__(16) bf16t Plds[ATTN_WAVES][16][32];

    const int lane = threadIdx.x & 31;
    const int wib  = threadIdx.x >> 5;
    const int wave = blockIdx.x * ATTN_WAVES + wib;
    const int tilesPerB = N >> 4;
    if (wave >= Bn * tilesPerB) return;      // wave-uniform

    const int b   = wave / tilesPerB;
    const int it  = wave % tilesPerB;
    const int i0  = it << 4;
    const int l16 = lane & 15;
    const int hs  = lane >> 4;               // half-wave select

    const bf16t* qb = qT  + (size_t)b * N * 16;
    const bf16t* kb = kT  + (size_t)b * N * 16;
    const bf16t* vb = vbf + (size_t)b * C * N;
    const float* xb = x1  + (size_t)b * C * N;
    float*       fb = feat + (size_t)b * C * N;

    // Invariant A operand: q rows i0..i0+15, K=0..15 valid, K=16..31 zero.
    // A order used here (contiguous k, as B): lane<16 -> (M=l16, K 0..15); lane>=16 -> (M=l16, K 16..31).
    v16bf aq;
    {
        v8bf qlo = *(const v8bf*)(qb + (size_t)(i0 + l16) * 16);
        v8bf qhi = *(const v8bf*)(qb + (size_t)(i0 + l16) * 16 + 8);
        #pragma unroll
        for (int e = 0; e < 8; ++e) { aq[e] = hs == 0 ? qlo[e] : (bf16t)0.f; aq[e + 8] = hs == 0 ? qhi[e] : (bf16t)0.f; }
    }

    v8f zeroacc;
    #pragma unroll
    for (int e = 0; e < 8; ++e) zeroacc[e] = 0.f;

    float m[8], l[8];
    v8f O[4];
    #pragma unroll
    for (int r = 0; r < 8; ++r) { m[r] = -__builtin_inff(); l[r] = 0.f; }
    #pragma unroll
    for (int ct = 0; ct < 4; ++ct) O[ct] = zeroacc;

    const int irow = hs * 8;                 // row offset of this half-wave in C-layout

    for (int jb = 0; jb < N; jb += 32) {
        // ---- energy: two 16x16 j-subtiles -------------------------------
        v16bf bk0, bk1;
        #pragma unroll
        for (int e = 0; e < 16; ++e) { bk0[e] = (bf16t)0.f; bk1[e] = (bf16t)0.f; }
        if (hs == 0) {   // B, contiguous k as in A: lanes<16 carry K=0..15 (valid); lanes>=16 K=16..31 (zero pad)
            bk0 = *(const v16bf*)(kb + (size_t)(jb + l16) * 16);
            bk1 = *(const v16bf*)(kb + (size_t)(jb + 16 + l16) * 16);
        }
        v8f s0 = __builtin_amdgcn_wmma_f32_16x16x32_bf16(false, aq, false, bk0, (short)0, zeroacc, false, false);
        v8f s1 = __builtin_amdgcn_wmma_f32_16x16x32_bf16(false, aq, false, bk1, (short)0, zeroacc, false, false);

        // ---- online softmax (per-row, DPP 16-lane reductions) -----------
        float p0[8], p1[8], scl[8];
        #pragma unroll
        for (int r = 0; r < 8; ++r) {
            float bm = row16_max(fmaxf(s0[r], s1[r]));
            float mn = fmaxf(m[r], bm);
            float sc = __expf(m[r] - mn);
            p0[r] = __expf(s0[r] - mn);
            p1[r] = __expf(s1[r] - mn);
            float rs = row16_sum(p0[r] + p1[r]);
            l[r] = l[r] * sc + rs;
            m[r] = mn;
            scl[r] = sc;
        }
        #pragma unroll
        for (int ct = 0; ct < 4; ++ct)
            #pragma unroll
            for (int r = 0; r < 8; ++r) O[ct][r] *= scl[r];

        // ---- P: C-layout -> row-major LDS tile [16][32] ------------------
        #pragma unroll
        for (int r = 0; r < 8; ++r) {
            Plds[wib][irow + r][l16]      = (bf16t)p0[r];
            Plds[wib][irow + r][16 + l16] = (bf16t)p1[r];
        }
        asm volatile("s_wait_dscnt 0" ::: "memory");

        // ---- reload P in A layout ---------------------------------------
        v16bf ap;
        {
            const bf16t* pr = &Plds[wib][l16][0];
            v8bf plo = *(const v8bf*)(pr + hs * 16);
            v8bf phi = *(const v8bf*)(pr + hs * 16 + 8);
            #pragma unroll
            for (int e = 0; e < 8; ++e) { ap[e] = plo[e]; ap[e + 8] = phi[e]; }
        }

        // ---- O += P x vT, 4 c-tiles -------------------------------------
        #pragma unroll
        for (int ct = 0; ct < 4; ++ct) {
            const bf16t* vr = vb + (size_t)(ct * 16 + l16) * N + jb + hs * 16;
            v16bf bvv = *(const v16bf*)vr;
            O[ct] = __builtin_amdgcn_wmma_f32_16x16x32_bf16(false, ap, false, bvv, (short)0, O[ct], false, false);
        }
        asm volatile("s_wait_dscnt 0" ::: "memory");  // P reads done before next overwrite
    }

    // ---- epilogue: normalize, gamma * out + residual ---------------------
    const float g = gamma[0];
    float inv[8];
    #pragma unroll
    for (int r = 0; r < 8; ++r) inv[r] = 1.0f / l[r];
    #pragma unroll
    for (int ct = 0; ct < 4; ++ct) {
        int c = ct * 16 + l16;
        #pragma unroll
        for (int r = 0; r < 8; ++r) {
            int i = i0 + irow + r;
            size_t off = (size_t)c * N + i;
            fb[off] = fmaf(g, O[ct][r] * inv[r], xb[off]);
        }
    }
}

// ---------------------------------------------------------------------------
extern "C" void kernel_launch(void* const* d_in, const int* in_sizes, int n_in,
                              void* d_out, int out_size, void* d_ws, size_t ws_size,
                              hipStream_t stream)
{
    const float* x     = (const float*)d_in[0];
    const float* w_pre = (const float*)d_in[1];
    const float* bn1g  = (const float*)d_in[2];
    const float* bn1b  = (const float*)d_in[3];
    const float* bn1m  = (const float*)d_in[4];
    const float* bn1v  = (const float*)d_in[5];
    const float* wq    = (const float*)d_in[6];
    const float* bq    = (const float*)d_in[7];
    const float* wk    = (const float*)d_in[8];
    const float* bk    = (const float*)d_in[9];
    const float* wv    = (const float*)d_in[10];
    const float* bv    = (const float*)d_in[11];
    const float* w_fin = (const float*)d_in[12];
    const float* bn2g  = (const float*)d_in[13];
    const float* bn2b  = (const float*)d_in[14];
    const float* bn2m  = (const float*)d_in[15];
    const float* bn2v  = (const float*)d_in[16];
    const float* gamma = (const float*)d_in[17];
    (void)in_sizes; (void)n_in; (void)out_size; (void)ws_size;

    const int B = 2, Cin = 64, C = 64, H = 96, W = 96, Cqk = 16;
    const int N = H * W;                       // 9216

    char* ws = (char*)d_ws;
    size_t off = 0;
    auto salloc = [&](size_t bytes) -> void* {
        void* p = ws + off;
        off = (off + bytes + 255) & ~(size_t)255;
        return p;
    };
    float* x1   = (float*)salloc((size_t)B * C * N * sizeof(float));
    float* feat = (float*)salloc((size_t)B * C * N * sizeof(float));
    bf16t* qT   = (bf16t*)salloc((size_t)B * N * Cqk * sizeof(bf16t));
    bf16t* kT   = (bf16t*)salloc((size_t)B * N * Cqk * sizeof(bf16t));
    bf16t* vbf  = (bf16t*)salloc((size_t)B * C * N * sizeof(bf16t));

    // 1) pre_conv 3x3 + BN + ReLU
    {
        int total = B * C * H * W;
        int blocks = (total + 255) / 256;
        conv3x3_bn_relu_kernel<<<blocks, 256, 0, stream>>>(
            x, w_pre, bn1g, bn1b, bn1m, bn1v, x1, B, Cin, C, H, W);
    }
    // 2) q/k/v 1x1 projections -> bf16 WMMA layouts
    {
        int total = B * (2 * Cqk + C) * N;
        int blocks = (total + 255) / 256;
        qkv_kernel<<<blocks, 256, 0, stream>>>(
            x1, wq, bq, wk, bk, wv, bv, qT, kT, vbf, B, C, Cqk, N);
    }
    // 3) flash attention + residual (WMMA core)
    {
        int tiles  = B * (N / 16);             // 1152 waves
        int blocks = (tiles + ATTN_WAVES - 1) / ATTN_WAVES;
        flash_attn_kernel<<<blocks, ATTN_WAVES * 32, 0, stream>>>(
            qT, kT, vbf, x1, gamma, feat, B, N, C);
    }
    // 4) final_conv 3x3 + BN + ReLU -> d_out
    {
        int total = B * C * H * W;
        int blocks = (total + 255) / 256;
        conv3x3_bn_relu_kernel<<<blocks, 256, 0, stream>>>(
            feat, w_fin, bn2g, bn2b, bn2m, bn2v, (float*)d_out, B, C, C, H, W);
    }
}
